// KPConv_71571335021213
// MI455X (gfx1250) — matrix operand order repaired and checked by reading, not compiled, not run on hardware
//
#include <hip/hip_runtime.h>
#include <hip/hip_bf16.h>

typedef __attribute__((ext_vector_type(16))) __bf16 v16bf;
typedef __attribute__((ext_vector_type(8)))  float  v8f;

#define N_Q     40000
#define N_S     40000
#define K_NB    32
#define P_KP    15
#define IN_DIM  64
#define OUT_DIM 128
#define KP_EXT  0.05f

union Frag { v16bf v; uint4 q[2]; };

// round-to-nearest-even f32 -> bf16 bits (used only in one-time weight pack)
__device__ __forceinline__ unsigned short f2bf_rne(float f) {
    unsigned int u = __float_as_uint(f);
    u += 0x7fffu + ((u >> 16) & 1u);
    return (unsigned short)(u >> 16);
}

// truncating f32 -> bf16 bits: compiles to a plain d16_hi store (no ALU)
__device__ __forceinline__ unsigned short f2bf_trunc(float f) {
    return (unsigned short)(__float_as_uint(f) >> 16);
}

// pack two f32 into {bf16(hi), bf16(lo)} dword (truncating)
__device__ __forceinline__ unsigned int pack_bf2(float lo, float hi) {
    return (__float_as_uint(hi) & 0xffff0000u) | (__float_as_uint(lo) >> 16);
}

// Inverse of the 16-bit A-matrix (16x32) fragment layout: given k in [0,32),
// produce lane-half (0/1) and the element index e in the v16 fragment.
// Forward map (contiguous order, matching the B operands): K = half*16 + e.
__device__ __forceinline__ void a_slot(int k, int& half, int& e) {
    half = (k >> 4) & 1;
    e = k & 15;
}

// ---------------------------------------------------------------------------
// Pack W_flat[1024x128] (K = p*64+d, zero-padded past 960) as bf16 fragments.
// Layout: block b = kTile*8 + nTile; within a block lane L holds 16 contiguous
// bf16: B[K = kTile*32 + 16*(L/16) + e][N = nTile*16 + L%16].
// ---------------------------------------------------------------------------
__global__ void pack_weights_kernel(const float* __restrict__ w,
                                    unsigned short* __restrict__ wb) {
    int gid = blockIdx.x * blockDim.x + threadIdx.x;   // 0 .. 131071
    int e    = gid & 15;
    int lane = (gid >> 4) & 31;
    int b    = gid >> 9;
    int kTile = b >> 3, nTile = b & 7;
    int Kr = kTile * 32 + ((lane >> 4) << 4) + e;
    int Nc = nTile * 16 + (lane & 15);
    float v = (Kr < P_KP * IN_DIM) ? w[(size_t)Kr * OUT_DIM + Nc] : 0.0f;
    wb[gid] = f2bf_rne(v);
}

// ---------------------------------------------------------------------------
// Main fused kernel: one block = 16 queries, 256 threads = 8 wave32s.
// ---------------------------------------------------------------------------
__global__ void __launch_bounds__(256)
kpconv_main_kernel(const float* __restrict__ q_pts,
                   const float* __restrict__ s_pts,
                   const int*   __restrict__ nbr,
                   const float* __restrict__ x,
                   const unsigned short* __restrict__ wb_packed,
                   const float* __restrict__ kpts,
                   float* __restrict__ out) {
    extern __shared__ char smem[];
    // nxS: neighbor features bf16, [q][d][k]  (16*64*32 shorts = 64 KB)
    unsigned short* nxS = (unsigned short*)smem;
    // awS: correlation A-fragments, per query 32 lanes x 16 e (16*512 = 16 KB)
    unsigned short* awS = nxS + 16 * IN_DIM * K_NB;
    // wfS: weighted-feature A-fragments for phase B, 32 kTiles x 512 (32 KB)
    unsigned short* wfS = awS + 16 * 512;
    // centered neighbor coords f32 [q][k][3] (6 KB)
    float* coS = (float*)(wfS + 32 * 512);
    // kernel points (45 floats, padded)
    float* kpS = coS + 16 * K_NB * 3;
    // valid-neighbor counts
    int* cntS = (int*)(kpS + 48);

    const int tid = threadIdx.x;
    const int q0  = blockIdx.x * 16;

    // ---- stage kernel points, init counters ----
    if (tid < P_KP * 3) kpS[tid] = kpts[tid];
    if (tid < 16) cntS[tid] = 0;
    __syncthreads();

    // ---- gather: each thread owns one (q, adjacent-k-pair) ----
    {
        int q = tid >> 4;
        int k = (tid & 15) * 2;
        int idx0 = nbr[(size_t)(q0 + q) * K_NB + k + 0];
        int idx1 = nbr[(size_t)(q0 + q) * K_NB + k + 1];
        bool sh0 = (idx0 >= N_S) || (idx0 < 0);
        bool sh1 = (idx1 >= N_S) || (idx1 < 0);
        float qx = q_pts[(q0 + q) * 3 + 0];
        float qy = q_pts[(q0 + q) * 3 + 1];
        float qz = q_pts[(q0 + q) * 3 + 2];
        float s0x = 0.f, s0y = 0.f, s0z = 0.f;
        float s1x = 0.f, s1y = 0.f, s1z = 0.f;
        if (!sh0) { s0x = s_pts[(size_t)idx0 * 3 + 0];
                    s0y = s_pts[(size_t)idx0 * 3 + 1];
                    s0z = s_pts[(size_t)idx0 * 3 + 2]; }
        if (!sh1) { s1x = s_pts[(size_t)idx1 * 3 + 0];
                    s1y = s_pts[(size_t)idx1 * 3 + 1];
                    s1z = s_pts[(size_t)idx1 * 3 + 2]; }
        coS[(q * K_NB + k + 0) * 3 + 0] = s0x - qx;
        coS[(q * K_NB + k + 0) * 3 + 1] = s0y - qy;
        coS[(q * K_NB + k + 0) * 3 + 2] = s0z - qz;
        coS[(q * K_NB + k + 1) * 3 + 0] = s1x - qx;
        coS[(q * K_NB + k + 1) * 3 + 1] = s1y - qy;
        coS[(q * K_NB + k + 1) * 3 + 2] = s1z - qz;

        const float4* xr0 = (const float4*)(x + (size_t)idx0 * IN_DIM);
        const float4* xr1 = (const float4*)(x + (size_t)idx1 * IN_DIM);
        // packed dword store covers (k, k+1) of a given d: 4-byte aligned
        unsigned int* dst = (unsigned int*)(nxS + q * (IN_DIM * K_NB) + k);
        float sum0 = 0.f, sum1 = 0.f;
        #pragma unroll
        for (int d4 = 0; d4 < IN_DIM / 4; ++d4) {
            float4 v0 = sh0 ? make_float4(0.f, 0.f, 0.f, 0.f) : xr0[d4];
            float4 v1 = sh1 ? make_float4(0.f, 0.f, 0.f, 0.f) : xr1[d4];
            sum0 += v0.x + v0.y + v0.z + v0.w;
            sum1 += v1.x + v1.y + v1.z + v1.w;
            int d = d4 * 4;
            dst[(d + 0) * (K_NB / 2)] = pack_bf2(v0.x, v1.x);
            dst[(d + 1) * (K_NB / 2)] = pack_bf2(v0.y, v1.y);
            dst[(d + 2) * (K_NB / 2)] = pack_bf2(v0.z, v1.z);
            dst[(d + 3) * (K_NB / 2)] = pack_bf2(v0.w, v1.w);
        }
        if (sum0 > 0.f) atomicAdd(&cntS[q], 1);
        if (sum1 > 0.f) atomicAdd(&cntS[q], 1);
    }
    __syncthreads();

    // ---- kernel-point correlations straight into A-fragment layout ----
    // A matrix per query: M = p (row 15 zero), K = k.
    for (int i = tid; i < 16 * 16 * K_NB; i += 256) {
        int q = i >> 9, p = (i >> 5) & 15, k = i & 31;
        float v = 0.f;
        if (p < P_KP) {
            float dx = coS[(q * K_NB + k) * 3 + 0] - kpS[p * 3 + 0];
            float dy = coS[(q * K_NB + k) * 3 + 1] - kpS[p * 3 + 1];
            float dz = coS[(q * K_NB + k) * 3 + 2] - kpS[p * 3 + 2];
            float dist = sqrtf(dx * dx + dy * dy + dz * dz);
            v = fmaxf(0.f, 1.f - dist * (1.0f / KP_EXT));
        }
        int half, e; a_slot(k, half, e);
        awS[q * 512 + (half * 16 + p) * 16 + e] = f2bf_trunc(v);
    }
    __syncthreads();

    const int lane = tid & 31;
    const int wv   = tid >> 5;

    // ---- phase A: per query  wf[16p x 64d] = aw[16x32] @ nx[32x64] ----
    // wave w handles queries 2w, 2w+1; 4 WMMAs per query (dim chunks of 16).
    for (int qi = 0; qi < 2; ++qi) {
        int q = wv * 2 + qi;
        Frag A;
        const uint4* ap = (const uint4*)(awS + q * 512 + lane * 16);
        A.q[0] = ap[0]; A.q[1] = ap[1];
        #pragma unroll
        for (int c = 0; c < 4; ++c) {
            Frag B;
            const uint4* bp = (const uint4*)(nxS + q * (IN_DIM * K_NB)
                                             + (c * 16 + (lane & 15)) * K_NB
                                             + ((lane >> 4) << 4));
            B.q[0] = bp[0]; B.q[1] = bp[1];
            v8f acc = {};
            acc = __builtin_amdgcn_wmma_f32_16x16x32_bf16(
                false, A.v, false, B.v, (short)0, acc, false, false);
            // scatter D into phase-B A-fragment layout (M'=q, K'=p*64+d).
            // Truncating stores -> ds_store_b16_d16_hi straight from acc.
            int d = c * 16 + (lane & 15);
            int hbase = (lane >> 4) * 8;
            #pragma unroll
            for (int r = 0; r < 8; ++r) {
                int p  = r + hbase;
                int Kp = p * IN_DIM + d;
                int t = Kp >> 5, k32 = Kp & 31;
                int half, e; a_slot(k32, half, e);
                wfS[t * 512 + (half * 16 + q) * 16 + e] = f2bf_trunc(acc[r]);
            }
        }
    }
    __syncthreads();

    // ---- phase B: out[16 x 128] = wf[16 x 1024] @ Wb[1024 x 128] ----
    // wave w owns columns [16w, 16w+16); dual accumulators break the
    // WMMA->WMMA RAW chain on C so the matrix pipe can stay full.
    const uint4* wb4 = (const uint4*)wb_packed;
    v8f acc0 = {}, acc1 = {};
    for (int t = 0; t < 32; t += 2) {
        Frag A0, B0, A1, B1;
        const uint4* ap0 = (const uint4*)(wfS + (t + 0) * 512 + lane * 16);
        const uint4* ap1 = (const uint4*)(wfS + (t + 1) * 512 + lane * 16);
        A0.q[0] = ap0[0]; A0.q[1] = ap0[1];
        A1.q[0] = ap1[0]; A1.q[1] = ap1[1];
        int b0 = (t + 0) * 8 + wv;
        int b1 = (t + 1) * 8 + wv;
        B0.q[0] = wb4[b0 * 64 + lane * 2 + 0];
        B0.q[1] = wb4[b0 * 64 + lane * 2 + 1];
        B1.q[0] = wb4[b1 * 64 + lane * 2 + 0];
        B1.q[1] = wb4[b1 * 64 + lane * 2 + 1];
        if (t + 2 < 32)
            __builtin_prefetch(&wb4[(b0 + 16) * 64 + lane * 2], 0, 0);
        acc0 = __builtin_amdgcn_wmma_f32_16x16x32_bf16(
            false, A0.v, false, B0.v, (short)0, acc0, false, false);
        acc1 = __builtin_amdgcn_wmma_f32_16x16x32_bf16(
            false, A1.v, false, B1.v, (short)0, acc1, false, false);
    }

    int col   = wv * 16 + (lane & 15);
    int rbase = (lane >> 4) * 8;
    #pragma unroll
    for (int r = 0; r < 8; ++r) {
        int qq = r + rbase;
        int c  = cntS[qq];
        float inv = 1.0f / (float)(c > 0 ? c : 1);
        out[(size_t)(q0 + qq) * OUT_DIM + col] = (acc0[r] + acc1[r]) * inv;
    }
}

extern "C" void kernel_launch(void* const* d_in, const int* in_sizes, int n_in,
                              void* d_out, int out_size, void* d_ws, size_t ws_size,
                              hipStream_t stream) {
    const float* q_pts = (const float*)d_in[0];
    const float* s_pts = (const float*)d_in[1];
    const int*   nbr   = (const int*)  d_in[2];
    const float* x     = (const float*)d_in[3];
    const float* w     = (const float*)d_in[4];
    const float* kp    = (const float*)d_in[5];
    float* out = (float*)d_out;

    unsigned short* wb = (unsigned short*)d_ws;   // 32*8*32*16 bf16 = 256 KB

    // pack weights into WMMA B-fragment order (131072 elements)
    pack_weights_kernel<<<512, 256, 0, stream>>>(w, wb);

    size_t smem_bytes =
        (size_t)(16 * IN_DIM * K_NB + 16 * 512 + 32 * 512) * sizeof(unsigned short)
        + (size_t)(16 * K_NB * 3 + 48) * sizeof(float)
        + 16 * sizeof(int);

    kpconv_main_kernel<<<N_Q / 16, 256, smem_bytes, stream>>>(
        q_pts, s_pts, nbr, x, wb, kp, out);
}
